// SGNNMPNN_35983236006070
// MI455X (gfx1250) — matrix operand order repaired and checked by reading, not compiled, not run on hardware
//
#include <hip/hip_runtime.h>
#include <hip/hip_fp16.h>
#include <math.h>

#define HW_N   65536
#define CH     200
#define NSP    1024
#define NE     524288
#define DD     128
#define NCLS   16

typedef __attribute__((ext_vector_type(16))) _Float16 v16h;
typedef __attribute__((ext_vector_type(8)))  float    v8f;

union FragH { unsigned u[8]; v16h v; };

__device__ __forceinline__ float leakyf(float x){ return x >= 0.f ? x : 0.01f * x; }

__device__ __forceinline__ unsigned pack2h(float a, float b){
  __half2 h = __float22half2_rn(make_float2(a, b));   // one v_cvt_pk_f16_f32
  return __builtin_bit_cast(unsigned, h);
}

// ---------------- utility: zero a float buffer (graph-capture safe) ----------------
__global__ void k_zero(float* __restrict__ p, int n){
  int i = blockIdx.x * blockDim.x + threadIdx.x;
  int stride = gridDim.x * blockDim.x;
  for (; i < n; i += stride) p[i] = 0.f;
}

// ---------------- seg[p] = argmax(one-hot Q row) ----------------
__global__ __launch_bounds__(256) void k_seg(const float* __restrict__ Q, int* __restrict__ seg){
  int w    = (blockIdx.x * blockDim.x + threadIdx.x) >> 5;
  int lane = threadIdx.x & 31;
  if (w >= HW_N) return;
  const float* row = Q + (size_t)w * NSP;
  #pragma unroll
  for (int j = 0; j < 32; ++j){
    int c = lane + (j << 5);
    if (row[c] > 0.5f) seg[w] = c;   // exactly one lane/iter hits (one-hot)
  }
}

// ---------------- counts ----------------
__global__ void k_cnt_sp(const int* __restrict__ seg, float* __restrict__ cnt){
  int p = blockIdx.x * blockDim.x + threadIdx.x;
  if (p < HW_N) atomicAdd(&cnt[seg[p]], 1.0f);
}

__global__ void k_deg(const int* __restrict__ ei, float* __restrict__ cnt){
  int e = blockIdx.x * blockDim.x + threadIdx.x;
  if (e < NE) atomicAdd(&cnt[ei[NE + e]], 1.0f);
}

__global__ void k_dinv(const float* __restrict__ cnt, float* __restrict__ dinvv, float* __restrict__ invdeg){
  int i = blockIdx.x * blockDim.x + threadIdx.x;
  if (i >= HW_N) return;
  float d = cnt[i] + 1.0f;            // + self loop
  dinvv[i]  = rsqrtf(d);
  invdeg[i] = 1.0f / d;
}

// ---------------- column stats (sum / sumsq) over [M,128] ----------------
__global__ void k_colstats(const float* __restrict__ X, float* __restrict__ sums, int M, int rowsPerBlock){
  int c  = threadIdx.x;               // 0..127
  int r0 = blockIdx.x * rowsPerBlock;
  int r1 = r0 + rowsPerBlock; if (r1 > M) r1 = M;
  float s = 0.f, q = 0.f;
  for (int r = r0; r < r1; ++r){ float v = X[(size_t)r * DD + c]; s += v; q += v * v; }
  atomicAdd(&sums[c], s);
  atomicAdd(&sums[DD + c], q);
}

__global__ void k_finstats(const float* __restrict__ sums, float* __restrict__ mean,
                           float* __restrict__ rstd, float invM){
  int c = threadIdx.x;
  float m = sums[c] * invM;
  float v = sums[DD + c] * invM - m * m;
  if (v < 0.f) v = 0.f;
  mean[c] = m;
  rstd[c] = rsqrtf(v + 1e-5f);
}

// ---- fold BN shift into per-column bias: c0[n] = -sum_k mean[k]*rstd[k]*W[k][n] ----
__global__ void k_fold_bias(const float* __restrict__ W, const float* __restrict__ mean,
                            const float* __restrict__ rstd, float* __restrict__ c0){
  __shared__ float ms[DD];
  int n = threadIdx.x;
  ms[n] = mean[n] * rstd[n];
  __syncthreads();
  float acc = 0.f;
  for (int k = 0; k < DD; ++k) acc += ms[k] * W[k * DD + n];
  c0[n] = -acc;
}

// ---------------- apply BN + superpixel pooling scatter ----------------
__global__ void k_bn_pool(const float* __restrict__ X, const float* __restrict__ mean,
                          const float* __restrict__ rstd, const int* __restrict__ seg,
                          float* __restrict__ H, float* __restrict__ sp_sum){
  int idx = blockIdx.x * blockDim.x + threadIdx.x;   // HW_N*DD total
  int r = idx >> 7, c = idx & 127;
  float v = (X[idx] - mean[c]) * rstd[c];
  H[idx] = v;
  atomicAdd(&sp_sum[seg[r] * DD + c], v);
}

__global__ void k_sp_mean(const float* __restrict__ ssum, const float* __restrict__ cnt, float* __restrict__ sp){
  int idx = blockIdx.x * blockDim.x + threadIdx.x;
  if (idx < NSP * DD) sp[idx] = ssum[idx] / cnt[idx >> 7];
}

// ---------------- row-wise L2 normalize [n,128], wave per row ----------------
__global__ __launch_bounds__(256) void k_l2norm_rows(const float* __restrict__ X, float* __restrict__ Y, int nrows){
  int w    = (blockIdx.x * blockDim.x + threadIdx.x) >> 5;
  int lane = threadIdx.x & 31;
  if (w >= nrows) return;
  const float4 x4 = ((const float4*)X)[(size_t)w * 32 + lane];
  float s = x4.x * x4.x + x4.y * x4.y + x4.z * x4.z + x4.w * x4.w;
  #pragma unroll
  for (int off = 16; off > 0; off >>= 1) s += __shfl_xor(s, off, 32);
  float inv = 1.0f / fmaxf(sqrtf(s), 1e-12f);
  float4 y4; y4.x = x4.x * inv; y4.y = x4.y * inv; y4.z = x4.z * inv; y4.w = x4.w * inv;
  ((float4*)Y)[(size_t)w * 32 + lane] = y4;
}

// ======================================================================
// WMMA f16 GEMM, f32 accumulate:  C[M,N] = A[M,K] @ diag(bscale) @ B[K,N] + bias
// (BatchNorm folded: scale into B staging, shift into bias via k_fold_bias.)
// Block: 256 thr (8 waves). Tile M=128, N=64, K=32. Double-buffered LDS with
// register prefetch; one barrier per K-step. TAIL=false -> no bounds checks.
// ======================================================================
#define GM 128
#define GN 64
#define GK 32
template<bool SCALEB, bool TAIL>
__global__ __launch_bounds__(256, 1)
void k_gemm_wmma(const float* __restrict__ A, const float* __restrict__ B,
                 const float* __restrict__ bias, const float* __restrict__ bscale,
                 float* __restrict__ C, int M, int N, int K)
{
  // LDS as packed f16 pairs (uint): As[r][kpair], Bs[n][kpair]
  __shared__ unsigned As[2][GM * (GK/2)];   // 2 x 8KB
  __shared__ unsigned Bs[2][GN * (GK/2)];   // 2 x 4KB
  int tid  = threadIdx.x;
  int wid  = tid >> 5;
  int lane = tid & 31;
  int r0   = blockIdx.x * GM;
  int nb0  = blockIdx.y * GN;

  // staging coordinates: A = 4 float4/thread, B = 2 float4/thread (k, k+1 rows)
  int a_r[4], a_k[4];
  #pragma unroll
  for (int i = 0; i < 4; ++i){ int idx = tid + (i << 8); a_r[i] = idx >> 3; a_k[i] = (idx & 7) << 2; }
  const int b_kp = tid >> 4;          // 0..15  (k pair)
  const int b_n  = (tid & 15) << 2;   // 0..60  (n group of 4)

  float4 aReg[4], b0Reg, b1Reg;
  float2 rsReg;                       // bscale pair for this thread's k-pair

  auto loadTile = [&](int kt){
    int kc = kt * GK;
    #pragma unroll
    for (int i = 0; i < 4; ++i){
      int kg = kc + a_k[i];
      if (!TAIL || kg < K)            // K % 4 == 0, per-float4 guard is exact
        aReg[i] = *(const float4*)(A + (size_t)(r0 + a_r[i]) * K + kg);
      else
        aReg[i] = make_float4(0.f, 0.f, 0.f, 0.f);
    }
    int kg0 = kc + (b_kp << 1);
    if (!TAIL || kg0 < K){            // K % 2 == 0
      b0Reg = *(const float4*)(B + (size_t)kg0 * N + nb0 + b_n);
      b1Reg = *(const float4*)(B + (size_t)(kg0 + 1) * N + nb0 + b_n);
      if (SCALEB) rsReg = *(const float2*)(bscale + kg0);
    } else {
      b0Reg = make_float4(0.f, 0.f, 0.f, 0.f);
      b1Reg = b0Reg;
      if (SCALEB) rsReg = make_float2(1.f, 1.f);
    }
  };
  auto storeTile = [&](int p){
    #pragma unroll
    for (int i = 0; i < 4; ++i){
      float4 v = aReg[i];
      uint2 pk; pk.x = pack2h(v.x, v.y); pk.y = pack2h(v.z, v.w);
      *(uint2*)&As[p][a_r[i] * 16 + (a_k[i] >> 1)] = pk;   // 8B aligned (a_k%4==0)
    }
    const float* b0 = (const float*)&b0Reg;
    const float* b1 = (const float*)&b1Reg;
    #pragma unroll
    for (int i = 0; i < 4; ++i){
      float x0 = b0[i], x1 = b1[i];
      if (SCALEB){ x0 *= rsReg.x; x1 *= rsReg.y; }
      Bs[p][(b_n + i) * 16 + b_kp] = pack2h(x0, x1);
    }
  };

  v8f acc[4] = {};
  int nK = (K + GK - 1) / GK;

  loadTile(0);
  storeTile(0);
  __syncthreads();

  for (int kt = 0; kt < nK; ++kt){
    int p = kt & 1;
    if (kt + 1 < nK) loadTile(kt + 1);          // prefetch next tile into regs

    // A fragment (16x32 f16): lanes 0-15 K 0..7 & 16..23, lanes 16-31 K 8..15 & 24..31
    FragH fa;
    {
      int m  = (wid << 4) + (lane & 15);
      int ub = ((lane >> 4) << 2);
      const unsigned* Ar = &As[p][m * 16];
      #pragma unroll
      for (int j = 0; j < 4; ++j){ fa.u[j] = Ar[ub + j]; fa.u[4 + j] = Ar[8 + ub + j]; }
    }
    #pragma unroll
    for (int nt = 0; nt < 4; ++nt){
      FragH fb;
      int n   = (nt << 4) + (lane & 15);
      int ub2 = ((lane >> 4) << 2);             // lanes 0-15: K 0..7 & 16..23, lanes 16-31: K 8..15 & 24..31
      const unsigned* Br = &Bs[p][n * 16];
      #pragma unroll
      for (int j = 0; j < 4; ++j){ fb.u[j] = Br[ub2 + j]; fb.u[4 + j] = Br[8 + ub2 + j]; }
      acc[nt] = __builtin_amdgcn_wmma_f32_16x16x32_f16(
          false, fa.v, false, fb.v, (short)0, acc[nt], false, false);
    }

    if (kt + 1 < nK) storeTile(1 - p);
    __syncthreads();
  }

  // store C: lanes 0-15 -> M=r, lanes 16-31 -> M=r+8; N = lane&15
  int mb = r0 + (wid << 4) + ((lane >> 4) << 3);
  #pragma unroll
  for (int nt = 0; nt < 4; ++nt){
    int ng = nb0 + (nt << 4) + (lane & 15);
    float bv = bias ? bias[ng] : 0.f;
    #pragma unroll
    for (int r = 0; r < 8; ++r)
      C[(size_t)(mb + r) * N + ng] = acc[nt][r] + bv;
  }
}

// ---------------- small: Y[1024,128] = bn(T) @ W + b ----------------
__global__ void k_rowmat_bn(const float* __restrict__ T, const float* __restrict__ W,
                            const float* __restrict__ bias, const float* __restrict__ mean,
                            const float* __restrict__ rstd, float* __restrict__ Y){
  __shared__ float t[DD];
  int r = blockIdx.x, c = threadIdx.x;
  t[c] = (T[(size_t)r * DD + c] - mean[c]) * rstd[c];
  __syncthreads();
  float acc = bias[c];
  for (int k = 0; k < DD; ++k) acc += t[k] * W[k * DD + c];
  Y[(size_t)r * DD + c] = acc;
}

// ---------------- small: Z[1024,128] = leaky(A_norm[1024,1024] @ Y) ----------------
__global__ void k_spgemmA(const float* __restrict__ A, const float* __restrict__ Y, float* __restrict__ Z){
  __shared__ float a[NSP];
  int r = blockIdx.x, c = threadIdx.x;
  for (int i = c; i < NSP; i += DD) a[i] = A[(size_t)r * NSP + i];
  __syncthreads();
  float acc = 0.f;
  for (int k = 0; k < NSP; ++k) acc += a[k] * Y[k * DD + c];
  Z[(size_t)r * DD + c] = leakyf(acc);
}

// ---------------- edge scatter: wave per edge, float4 gather + 4 atomics/lane ----------------
__global__ __launch_bounds__(256) void k_edge_scatter(const int* __restrict__ ei,
    const float* __restrict__ dinvv, const float* __restrict__ xw, float* __restrict__ agg){
  int w    = (blockIdx.x * blockDim.x + threadIdx.x) >> 5;
  int lane = threadIdx.x & 31;
  if (w >= NE) return;
  int s = ei[w], d = ei[NE + w];
  float nrm = dinvv[s] * dinvv[d];
  const float4 v = ((const float4*)xw)[(size_t)s * 32 + lane];
  float* a = agg + (size_t)d * DD + (lane << 2);
  atomicAdd(a + 0, v.x * nrm);
  atomicAdd(a + 1, v.y * nrm);
  atomicAdd(a + 2, v.z * nrm);
  atomicAdd(a + 3, v.w * nrm);
}

// ---------------- combine: out = leaky(agg + xw/deg + b) ----------------
__global__ void k_combine(const float* __restrict__ agg, const float* __restrict__ xw,
    const float* __restrict__ invdeg, const float* __restrict__ b, float* __restrict__ out){
  int idx = blockIdx.x * blockDim.x + threadIdx.x;  // HW_N*DD
  int r = idx >> 7, c = idx & 127;
  out[idx] = leakyf(agg[idx] + xw[idx] * invdeg[r] + b[c]);
}

// ---------------- output: gather + GEMV(128x16) + softmax ----------------
__global__ __launch_bounds__(256) void k_output(const float* __restrict__ Mb, const float* __restrict__ H1,
     const int* __restrict__ seg, const float* __restrict__ Wo, const float* __restrict__ bo,
     float* __restrict__ out){
  __shared__ float Wl[DD * NCLS];
  __shared__ float bl[NCLS];
  int tid = threadIdx.x;
  for (int i = tid; i < DD * NCLS; i += 256) Wl[i] = Wo[i];
  if (tid < NCLS) bl[tid] = bo[tid];
  __syncthreads();
  int p = blockIdx.x * 256 + tid;
  const float4* mrow = (const float4*)(Mb + (size_t)p * DD);
  const float4* grow = (const float4*)(H1 + (size_t)seg[p] * DD);
  float acc[NCLS];
  #pragma unroll
  for (int j = 0; j < NCLS; ++j) acc[j] = bl[j];
  for (int k4 = 0; k4 < DD / 4; ++k4){
    float4 mv = mrow[k4], gv = grow[k4];
    float v[4] = { mv.x + gv.x, mv.y + gv.y, mv.z + gv.z, mv.w + gv.w };
    #pragma unroll
    for (int u = 0; u < 4; ++u){
      const float* wr = &Wl[(k4 * 4 + u) * NCLS];
      #pragma unroll
      for (int j = 0; j < NCLS; ++j) acc[j] += v[u] * wr[j];  // LDS broadcast (uniform k)
    }
  }
  float mx = acc[0];
  #pragma unroll
  for (int j = 1; j < NCLS; ++j) mx = fmaxf(mx, acc[j]);
  float s = 0.f;
  #pragma unroll
  for (int j = 0; j < NCLS; ++j){ acc[j] = expf(acc[j] - mx); s += acc[j]; }
  float inv = 1.f / s;
  float* orow = out + (size_t)p * NCLS;
  #pragma unroll
  for (int j = 0; j < NCLS; ++j) orow[j] = acc[j] * inv;
}

// ======================================================================
extern "C" void kernel_launch(void* const* d_in, const int* in_sizes, int n_in,
                              void* d_out, int out_size, void* d_ws, size_t ws_size,
                              hipStream_t stream) {
  (void)in_sizes; (void)n_in; (void)out_size; (void)ws_size;
  const float* x        = (const float*)d_in[0];
  const float* Q        = (const float*)d_in[1];
  const float* A_norm   = (const float*)d_in[2];
  const int*   ei       = (const int*)  d_in[3];
  const float* prelin_W = (const float*)d_in[4];
  const float* prelin_b = (const float*)d_in[5];
  const float* gcn1_W   = (const float*)d_in[6];
  const float* gcn1_b   = (const float*)d_in[7];
  const float* gcn2_W   = (const float*)d_in[8];
  const float* gcn2_b   = (const float*)d_in[9];
  const float* mpnn1_W  = (const float*)d_in[10];
  const float* mpnn1_b  = (const float*)d_in[11];
  const float* mpnn2_W  = (const float*)d_in[12];
  const float* mpnn2_b  = (const float*)d_in[13];
  const float* out_W    = (const float*)d_in[14];
  const float* out_b    = (const float*)d_in[15];
  float* out = (float*)d_out;

  // ---- workspace carve (floats) ----
  float* ws = (float*)d_ws;
  size_t off = 0;
  auto carve = [&](size_t n){ float* p = ws + off; off += n; return p; };
  float* buf0   = carve((size_t)HW_N * DD);  // l2norm scratch, then reused as agg
  float* buf1   = carve((size_t)HW_N * DD);  // h / mpnn layer io
  float* buf2   = carve((size_t)HW_N * DD);  // xw
  float* sp_sum = carve((size_t)NSP * DD);
  float* St     = carve((size_t)NSP * DD);
  float* Sy     = carve((size_t)NSP * DD);
  float* S0     = carve((size_t)NSP * DD);
  float* S1     = carve((size_t)NSP * DD);
  float* S2     = carve((size_t)NSP * DD);
  float* cnt_sp = carve(NSP);
  float* cntdeg = carve(HW_N);
  float* dinvv  = carve(HW_N);
  float* invdeg = carve(HW_N);
  float* sums   = carve(2 * DD);
  float* meanv  = carve(DD);
  float* rstdv  = carve(DD);
  float* c0     = carve(DD);
  int*   seg    = (int*)(ws + off); off += HW_N;

  dim3 blk(256);

  // ---- graph structure ----
  k_seg<<<dim3(HW_N / 8), blk, 0, stream>>>(Q, seg);
  k_zero<<<dim3(8),   blk, 0, stream>>>(cnt_sp, NSP);
  k_zero<<<dim3(256), blk, 0, stream>>>(cntdeg, HW_N);
  k_zero<<<dim3(512), blk, 0, stream>>>(sp_sum, NSP * DD);
  k_cnt_sp<<<dim3(HW_N / 256), blk, 0, stream>>>(seg, cnt_sp);
  k_deg<<<dim3(NE / 256), blk, 0, stream>>>(ei, cntdeg);
  k_dinv<<<dim3(HW_N / 256), blk, 0, stream>>>(cntdeg, dinvv, invdeg);

  // ---- prelin GEMM + BN + pooling ----
  k_gemm_wmma<false, true><<<dim3(HW_N / GM, DD / GN), blk, 0, stream>>>(
      x, prelin_W, prelin_b, nullptr, buf0, HW_N, DD, CH);
  k_zero<<<dim3(1), blk, 0, stream>>>(sums, 2 * DD);
  k_colstats<<<dim3(256), dim3(DD), 0, stream>>>(buf0, sums, HW_N, 256);
  k_finstats<<<dim3(1), dim3(DD), 0, stream>>>(sums, meanv, rstdv, 1.0f / (float)HW_N);
  k_bn_pool<<<dim3(HW_N * DD / 256), blk, 0, stream>>>(buf0, meanv, rstdv, seg, buf1, sp_sum);
  k_sp_mean<<<dim3(NSP * DD / 256), blk, 0, stream>>>(sp_sum, cnt_sp, S0);

  // ---- superpixel GCN layers ----
  auto gcn_layer = [&](const float* Sin, const float* W, const float* b, float* Sout){
    k_l2norm_rows<<<dim3(NSP / 8), blk, 0, stream>>>(Sin, St, NSP);
    k_zero<<<dim3(1), blk, 0, stream>>>(sums, 2 * DD);
    k_colstats<<<dim3(8), dim3(DD), 0, stream>>>(St, sums, NSP, 128);
    k_finstats<<<dim3(1), dim3(DD), 0, stream>>>(sums, meanv, rstdv, 1.0f / (float)NSP);
    k_rowmat_bn<<<dim3(NSP), dim3(DD), 0, stream>>>(St, W, b, meanv, rstdv, Sy);
    k_spgemmA<<<dim3(NSP), dim3(DD), 0, stream>>>(A_norm, Sy, Sout);
  };
  gcn_layer(S0, gcn1_W, gcn1_b, S1);
  gcn_layer(S1, gcn2_W, gcn2_b, S2);

  // ---- pixel MPNN layers (in/out = buf1) ----
  auto mpnn_layer = [&](const float* W, const float* b){
    k_l2norm_rows<<<dim3(HW_N / 8), blk, 0, stream>>>(buf1, buf0, HW_N);
    k_zero<<<dim3(1), blk, 0, stream>>>(sums, 2 * DD);
    k_colstats<<<dim3(256), dim3(DD), 0, stream>>>(buf0, sums, HW_N, 256);
    k_finstats<<<dim3(1), dim3(DD), 0, stream>>>(sums, meanv, rstdv, 1.0f / (float)HW_N);
    k_fold_bias<<<dim3(1), dim3(DD), 0, stream>>>(W, meanv, rstdv, c0);
    // xw = bn(t) @ W  ==  t @ (diag(rstd) W) + c0   (scale on B, shift as bias)
    k_gemm_wmma<true, false><<<dim3(HW_N / GM, DD / GN), blk, 0, stream>>>(
        buf0, W, c0, rstdv, buf2, HW_N, DD, DD);
    k_zero<<<dim3(4096), blk, 0, stream>>>(buf0, HW_N * DD);   // buf0 becomes agg
    k_edge_scatter<<<dim3(NE / 8), blk, 0, stream>>>(ei, dinvv, buf2, buf0);
    k_combine<<<dim3(HW_N * DD / 256), blk, 0, stream>>>(buf0, buf2, invdeg, b, buf1);
  };
  mpnn_layer(mpnn1_W, mpnn1_b);
  mpnn_layer(mpnn2_W, mpnn2_b);

  // ---- output: (m + H1[seg]) @ out_W + out_b, softmax ----
  k_output<<<dim3(HW_N / 256), blk, 0, stream>>>(buf1, S2, seg, out_W, out_b, out);
}
